// PosUpdate_83786222011177
// MI455X (gfx1250) — matrix operand order repaired and checked by reading, not compiled, not run on hardware
//
#include <hip/hip_runtime.h>
#include <math.h>

// ---------------------------------------------------------------------------
// CDNA5 (gfx1250) WMMA types and fragment helpers
// ---------------------------------------------------------------------------
typedef __attribute__((ext_vector_type(16))) __bf16 v16bf;
typedef __attribute__((ext_vector_type(8)))  float  v8f;

union FragB { uint4 q[2]; v16bf v; };
union Bf8   { uint4 q; __bf16 b[8]; };

#define N_NODES 50000
#define N_EDGES 800000
#define LDA 136   // padded bf16 row stride (272B/row -> 16B aligned, rotates banks)

// A fragment (16x32 bf16, M x K) from LDS, tile origin `base` (row M0, K=0), +k0.
// Lane L: m = L&15, h = L>>4. elem i: K = 16*(i/8) + 8h + (i%8)
//  -> two contiguous 8-elem (16B) chunks at K = 8h and K = 16+8h.
__device__ __forceinline__ v16bf load_a_lds(const __bf16* base, int lda, int k0) {
  int lane = threadIdx.x & 31;
  int m = lane & 15, h = lane >> 4;
  const __bf16* p = base + m * lda + k0 + h * 8;
  FragB f;
  f.q[0] = *(const uint4*)(p);
  f.q[1] = *(const uint4*)(p + 16);
  return f.v;
}

// B fragment (32x16 bf16, K x N) where B[k][n] = W[n][k] (W row-major, ldw = K).
// Lane L: n = L&15, h = L>>4. elem i: K = 16*(i/8) + 8h + (i%8) -> two 16B chunks at K = 8h and 16+8h.
__device__ __forceinline__ v16bf load_b_glb(const __bf16* __restrict__ W, int ldw,
                                            int n0, int k) {
  int lane = threadIdx.x & 31;
  int n = lane & 15, h = lane >> 4;
  const __bf16* p = W + (size_t)(n0 + n) * ldw + k + h * 8;
  FragB f;
  f.q[0] = *(const uint4*)(p);
  f.q[1] = *(const uint4*)(p + 16);
  return f.v;
}

// C += A(16x128) x W[n0:n0+16, kbase:kbase+128]^T  via 4x v_wmma_f32_16x16x32_bf16
__device__ __forceinline__ v8f mm128(const __bf16* A, int lda,
                                     const __bf16* __restrict__ W, int ldw,
                                     int n0, int kbase) {
  v8f c = {};
#pragma unroll
  for (int k0 = 0; k0 < 128; k0 += 32) {
    v16bf a = load_a_lds(A, lda, k0);
    v16bf b = load_b_glb(W, ldw, n0, kbase + k0);
    c = __builtin_amdgcn_wmma_f32_16x16x32_bf16(false, a, false, b,
                                                (short)0, c, false, false);
  }
  return c;
}

// Store C tile to LDS bf16 with bias (+optional relu). base includes M0*ld + n0.
__device__ __forceinline__ void st_lds_bf16(__bf16* base, int ld, v8f c,
                                            const float* bias, bool relu) {
  int lane = threadIdx.x & 31;
  int n = lane & 15, h = lane >> 4;
  float b = bias[n];
#pragma unroll
  for (int r = 0; r < 8; r++) {
    float v = c[r] + b;
    if (relu) v = v > 0.f ? v : 0.f;
    base[(r + 8 * h) * ld + n] = (__bf16)v;
  }
}

__device__ __forceinline__ void st_lds_f32(float* base, int ld, v8f c) {
  int lane = threadIdx.x & 31;
  int n = lane & 15, h = lane >> 4;
#pragma unroll
  for (int r = 0; r < 8; r++) base[(r + 8 * h) * ld + n] = c[r];
}

// ---------------------------------------------------------------------------
// Weight conversion: f32 (strided slice) -> packed bf16
// ---------------------------------------------------------------------------
__global__ void cvt_kernel(const float* __restrict__ src, __bf16* __restrict__ dst,
                           int rows, int srcLd, int colOff, int cols) {
  int i = blockIdx.x * blockDim.x + threadIdx.x;
  if (i < rows * cols) {
    int r = i / cols, c = i - r * cols;
    dst[(size_t)r * cols + c] = (__bf16)src[(size_t)r * srcLd + colOff + c];
  }
}

// ---------------------------------------------------------------------------
// Phase 1: per-node precompute.
//   left/right: f = relu(x@W1^T+b1)@W2^T+b2
//   nodebL[n] = f_l @ nodebW[:, :128]^T      (bf16)
//   nodebR[n] = f_r @ nodebW[:, 128:]^T      (bf16)
//   gateL[n]  = f_l @ gw1[:,128:256]^T + gate_b1 + node_extra @ gw1[:,384:386]^T
//   gateR[n]  = f_r @ gw1[:,256:384]^T
// 64 nodes/block, 128 threads (4 waves, one 16-row M-tile each).
// ---------------------------------------------------------------------------
__global__ __launch_bounds__(128) void node_pre_kernel(
    const float* __restrict__ h_node, const float* __restrict__ node_extra,
    const __bf16* __restrict__ lw1, const float* __restrict__ lb1,
    const __bf16* __restrict__ lw2, const float* __restrict__ lb2,
    const __bf16* __restrict__ rw1, const float* __restrict__ rb1,
    const __bf16* __restrict__ rw2, const float* __restrict__ rb2,
    const __bf16* __restrict__ nbW,
    const __bf16* __restrict__ g1l, const __bf16* __restrict__ g1r,
    const float* __restrict__ gate_w1f, const float* __restrict__ gate_b1,
    __bf16* __restrict__ nodebL, __bf16* __restrict__ nodebR,
    float* __restrict__ gateL, float* __restrict__ gateR) {
  __shared__ __bf16 Nb[64 * LDA];
  __shared__ __bf16 T1[64 * LDA];
  __shared__ __bf16 T2[64 * LDA];
  int tid = threadIdx.x;
  int base = blockIdx.x * 64;

  for (int i = tid; i < 64 * 32; i += 128) {  // h_node block -> bf16 LDS
    int row = i >> 5, c4 = i & 31;
    int g = base + row; if (g >= N_NODES) g = N_NODES - 1;
    float4 v = ((const float4*)(h_node + (size_t)g * 128))[c4];
    __bf16* p = Nb + row * LDA + c4 * 4;
    p[0] = (__bf16)v.x; p[1] = (__bf16)v.y; p[2] = (__bf16)v.z; p[3] = (__bf16)v.w;
  }
  __syncthreads();

  int w = tid >> 5, lane = tid & 31;
  int n = lane & 15, h = lane >> 4;
  int M0 = w * 16;

  float ne0[8], ne1[8]; int grow[8]; bool gok[8];
#pragma unroll
  for (int r = 0; r < 8; r++) {
    int g = base + M0 + 8 * h + r;
    gok[r] = (g < N_NODES); grow[r] = gok[r] ? g : 0;
    ne0[r] = node_extra[(size_t)grow[r] * 2 + 0];
    ne1[r] = node_extra[(size_t)grow[r] * 2 + 1];
  }

  for (int side = 0; side < 2; side++) {
    const __bf16* w1 = side ? rw1 : lw1; const float* b1 = side ? rb1 : lb1;
    const __bf16* w2 = side ? rw2 : lw2; const float* b2 = side ? rb2 : lb2;
    for (int t = 0; t < 8; t++) {  // hidden = relu(x@W1^T+b1)
      v8f c = mm128(Nb + M0 * LDA, LDA, w1, 128, t * 16, 0);
      st_lds_bf16(T1 + M0 * LDA + t * 16, LDA, c, b1 + t * 16, true);
    }
    for (int t = 0; t < 8; t++) {  // feat = hidden@W2^T+b2
      v8f c = mm128(T1 + M0 * LDA, LDA, w2, 128, t * 16, 0);
      st_lds_bf16(T2 + M0 * LDA + t * 16, LDA, c, b2 + t * 16, false);
    }
    __bf16* nout = side ? nodebR : nodebL;
    for (int t = 0; t < 8; t++) {  // nodeb projection (no bias)
      v8f c = mm128(T2 + M0 * LDA, LDA, nbW, 256, t * 16, side ? 128 : 0);
#pragma unroll
      for (int r = 0; r < 8; r++)
        if (gok[r]) nout[(size_t)grow[r] * 128 + t * 16 + n] = (__bf16)c[r];
    }
    float* gout = side ? gateR : gateL;
    const __bf16* gw = side ? g1r : g1l;
    for (int t = 0; t < 2; t++) {  // gate projection (32 wide)
      v8f c = mm128(T2 + M0 * LDA, LDA, gw, 128, t * 16, 0);
      int j = t * 16 + n;
      float gb = side ? 0.f : gate_b1[j];
      float x0 = side ? 0.f : gate_w1f[(size_t)j * 386 + 384];
      float x1 = side ? 0.f : gate_w1f[(size_t)j * 386 + 385];
#pragma unroll
      for (int r = 0; r < 8; r++)
        if (gok[r]) gout[(size_t)grow[r] * 32 + j] = c[r] + gb + ne0[r] * x0 + ne1[r] * x1;
    }
  }
}

// ---------------------------------------------------------------------------
// Phase 2: per-edge. 64 edges/block, 128 threads (4 waves x 16-edge M-tiles).
// ---------------------------------------------------------------------------
__global__ __launch_bounds__(128) void edge_kernel(
    const float* __restrict__ h_edge, const int* __restrict__ edge_index,
    const float* __restrict__ relative_vec, const float* __restrict__ distance,
    const __bf16* __restrict__ bondW, const __bf16* __restrict__ interW,
    const __bf16* __restrict__ g1e,
    const float* __restrict__ inter_b1, const float* __restrict__ inter_w2,
    const float* __restrict__ inter_b2,
    const float* __restrict__ gate_w2, const float* __restrict__ gate_b2,
    const __bf16* __restrict__ nodebL, const __bf16* __restrict__ nodebR,
    const float* __restrict__ gateL, const float* __restrict__ gateR,
    float* __restrict__ dpos) {
  __shared__ __bf16 Ae[64 * LDA];
  __shared__ float  Bond[64 * 132];
  __shared__ float  Gt[64 * 36];
  __shared__ int    elc[64], erc[64];
  __shared__ float  interS[64];
  int tid = threadIdx.x;
  long long base = (long long)blockIdx.x * 64;

  if (tid < 64) {
    elc[tid] = edge_index[base + tid];
    erc[tid] = edge_index[(long long)N_EDGES + base + tid];
  }
  for (int i = tid; i < 64 * 32; i += 128) {  // h_edge block -> bf16 LDS
    int row = i >> 5, c4 = i & 31;
    float4 v = ((const float4*)(h_edge + (base + row) * 128))[c4];
    __bf16* p = Ae + row * LDA + c4 * 4;
    p[0] = (__bf16)v.x; p[1] = (__bf16)v.y; p[2] = (__bf16)v.z; p[3] = (__bf16)v.w;
  }
  __syncthreads();

  int w = tid >> 5, lane = tid & 31;
  int n = lane & 15, h = lane >> 4;
  int M0 = w * 16;

  for (int t = 0; t < 2; t++) {  // gate edge part: h_edge @ gw1[:, :128]^T
    v8f c = mm128(Ae + M0 * LDA, LDA, g1e, 128, t * 16, 0);
    st_lds_f32(Gt + M0 * 36 + t * 16, 36, c);
  }
  for (int t = 0; t < 8; t++) {  // bond: h_edge @ bond_lin_w^T
    v8f c = mm128(Ae + M0 * LDA, LDA, bondW, 128, t * 16, 0);
    st_lds_f32(Bond + M0 * 132 + t * 16, 132, c);
  }

  {  // inter_in = bond + nodebL[el] + nodebR[er]  -> overwrite wave's Ae rows
    int e = M0 + n;
    int kb = h * 64;
    int el = elc[e], er = erc[e];
    const __bf16* nl = nodebL + (size_t)el * 128 + kb;
    const __bf16* nr = nodebR + (size_t)er * 128 + kb;
    const float* br = Bond + e * 132 + kb;
    __bf16* ar = Ae + e * LDA + kb;
#pragma unroll
    for (int k0 = 0; k0 < 64; k0 += 8) {
      Bf8 a, b;
      a.q = *(const uint4*)(nl + k0);
      b.q = *(const uint4*)(nr + k0);
#pragma unroll
      for (int j = 0; j < 8; j++)
        ar[k0 + j] = (__bf16)(br[k0 + j] + (float)a.b[j] + (float)b.b[j]);
    }
  }

  // inter MLP: relu(inter_in@W1^T+b1) . w2  (fused column reduction)
  float part[8];
#pragma unroll
  for (int r = 0; r < 8; r++) part[r] = 0.f;
  for (int t = 0; t < 8; t++) {
    v8f c = mm128(Ae + M0 * LDA, LDA, interW, 128, t * 16, 0);
    int col = t * 16 + n;
    float b1v = inter_b1[col], w2v = inter_w2[col];
#pragma unroll
    for (int r = 0; r < 8; r++) {
      float v = c[r] + b1v;
      v = v > 0.f ? v : 0.f;
      part[r] += v * w2v;
    }
  }
#pragma unroll
  for (int off = 1; off < 16; off <<= 1)
#pragma unroll
    for (int r = 0; r < 8; r++) part[r] += __shfl_xor(part[r], off, 32);
  if (n == 0) {
#pragma unroll
    for (int r = 0; r < 8; r++) interS[M0 + 8 * h + r] = part[r] + inter_b2[0];
  }

  if (lane < 16) {  // finalize one edge per lane
    int e = M0 + lane;
    int el = elc[e], er = erc[e];
    const float* gl = gateL + (size_t)el * 32;
    const float* gr = gateR + (size_t)er * 32;
    const float* ge = Gt + e * 36;
    float acc = 0.f;
#pragma unroll
    for (int j = 0; j < 32; j++) {
      float v = ge[j] + gl[j] + gr[j];
      v = v > 0.f ? v : 0.f;
      acc += v * gate_w2[j];
    }
    float sig = 1.f / (1.f + __expf(-(acc + gate_b2[0])));
    float wgt = interS[e] * sig;
    float d = distance[base + e];
    float fac = wgt * 5.f / ((d + 1e-6f) * (d + 5.f));
    long long g3 = (base + e) * 3;
    atomicAdd(&dpos[(size_t)el * 3 + 0], relative_vec[g3 + 0] * fac);
    atomicAdd(&dpos[(size_t)el * 3 + 1], relative_vec[g3 + 1] * fac);
    atomicAdd(&dpos[(size_t)el * 3 + 2], relative_vec[g3 + 2] * fac);
  }
}

// ---------------------------------------------------------------------------
// Phase 3: per-node scale + output.
// ---------------------------------------------------------------------------
__global__ __launch_bounds__(128) void finish_kernel(
    const float* __restrict__ h_node, const float* __restrict__ node_extra,
    const float* __restrict__ dpos,
    const __bf16* __restrict__ sw1h, const float* __restrict__ scale_w1f,
    const float* __restrict__ scale_b1, const float* __restrict__ scale_w2,
    const float* __restrict__ scale_b2, float* __restrict__ out) {
  __shared__ __bf16 Ab[64 * LDA];
  __shared__ float nrmB[64];
  __shared__ float dpB[64 * 3];
  int tid = threadIdx.x;
  int base = blockIdx.x * 64;

  for (int i = tid; i < 64 * 32; i += 128) {
    int row = i >> 5, c4 = i & 31;
    int g = base + row; if (g >= N_NODES) g = N_NODES - 1;
    float4 v = ((const float4*)(h_node + (size_t)g * 128))[c4];
    __bf16* p = Ab + row * LDA + c4 * 4;
    p[0] = (__bf16)v.x; p[1] = (__bf16)v.y; p[2] = (__bf16)v.z; p[3] = (__bf16)v.w;
  }
  if (tid < 64) {
    int g = base + tid; if (g >= N_NODES) g = N_NODES - 1;
    float dx = dpos[(size_t)g * 3 + 0], dy = dpos[(size_t)g * 3 + 1], dz = dpos[(size_t)g * 3 + 2];
    dpB[tid * 3 + 0] = dx; dpB[tid * 3 + 1] = dy; dpB[tid * 3 + 2] = dz;
    nrmB[tid] = sqrtf(dx * dx + dy * dy + dz * dz);
  }
  __syncthreads();

  int w = tid >> 5, lane = tid & 31;
  int n = lane & 15, h = lane >> 4;
  int M0 = w * 16;

  float ne0[8], ne1[8], nr[8]; int grow[8]; bool gok[8];
#pragma unroll
  for (int r = 0; r < 8; r++) {
    int row = M0 + 8 * h + r;
    int g = base + row;
    gok[r] = (g < N_NODES); grow[r] = gok[r] ? g : 0;
    ne0[r] = node_extra[(size_t)grow[r] * 2 + 0];
    ne1[r] = node_extra[(size_t)grow[r] * 2 + 1];
    nr[r] = nrmB[row];
  }

  float part[8];
#pragma unroll
  for (int r = 0; r < 8; r++) part[r] = 0.f;
  for (int t = 0; t < 8; t++) {
    v8f c = mm128(Ab + M0 * LDA, LDA, sw1h, 128, t * 16, 0);
    int col = t * 16 + n;
    float sb = scale_b1[col];
    float x0 = scale_w1f[(size_t)col * 131 + 128];
    float x1 = scale_w1f[(size_t)col * 131 + 129];
    float x2 = scale_w1f[(size_t)col * 131 + 130];
    float w2v = scale_w2[col];
#pragma unroll
    for (int r = 0; r < 8; r++) {
      float v = c[r] + sb + ne0[r] * x0 + ne1[r] * x1 + nr[r] * x2;
      v = v > 0.f ? v : 0.f;
      part[r] += v * w2v;
    }
  }
#pragma unroll
  for (int off = 1; off < 16; off <<= 1)
#pragma unroll
    for (int r = 0; r < 8; r++) part[r] += __shfl_xor(part[r], off, 32);

  if (n == 0) {
#pragma unroll
    for (int r = 0; r < 8; r++) {
      if (!gok[r]) continue;
      int row = M0 + 8 * h + r;
      float s = 1.f / (1.f + __expf(-(part[r] + scale_b2[0])));
      size_t g = (size_t)grow[r];
      out[g * 3 + 0] = dpB[row * 3 + 0] * s;
      out[g * 3 + 1] = dpB[row * 3 + 1] * s;
      out[g * 3 + 2] = dpB[row * 3 + 2] * s;
    }
  }
}

// ---------------------------------------------------------------------------
extern "C" void kernel_launch(void* const* d_in, const int* in_sizes, int n_in,
                              void* d_out, int out_size, void* d_ws, size_t ws_size,
                              hipStream_t stream) {
  const float* h_node       = (const float*)d_in[0];
  const float* h_edge       = (const float*)d_in[1];
  const int*   edge_index   = (const int*)d_in[2];
  const float* relative_vec = (const float*)d_in[3];
  const float* distance     = (const float*)d_in[4];
  const float* node_extra   = (const float*)d_in[5];
  const float* left_w1  = (const float*)d_in[6];
  const float* left_b1  = (const float*)d_in[7];
  const float* left_w2  = (const float*)d_in[8];
  const float* left_b2  = (const float*)d_in[9];
  const float* right_w1 = (const float*)d_in[10];
  const float* right_b1 = (const float*)d_in[11];
  const float* right_w2 = (const float*)d_in[12];
  const float* right_b2 = (const float*)d_in[13];
  const float* bond_lin_w  = (const float*)d_in[14];
  const float* nodeb_lin_w = (const float*)d_in[15];
  const float* inter_w1 = (const float*)d_in[16];
  const float* inter_b1 = (const float*)d_in[17];
  const float* inter_w2 = (const float*)d_in[18];
  const float* inter_b2 = (const float*)d_in[19];
  const float* gate_w1  = (const float*)d_in[20];
  const float* gate_b1  = (const float*)d_in[21];
  const float* gate_w2  = (const float*)d_in[22];
  const float* gate_b2  = (const float*)d_in[23];
  const float* scale_w1 = (const float*)d_in[24];
  const float* scale_b1 = (const float*)d_in[25];
  const float* scale_w2 = (const float*)d_in[26];
  const float* scale_b2 = (const float*)d_in[27];
  float* out = (float*)d_out;

  char* ws = (char*)d_ws;
  size_t off = 0;
  auto take = [&](size_t bytes) -> char* {
    char* p = ws + off;
    off = (off + bytes + 255) & ~(size_t)255;
    return p;
  };
  __bf16* nodebL = (__bf16*)take((size_t)N_NODES * 128 * 2);
  __bf16* nodebR = (__bf16*)take((size_t)N_NODES * 128 * 2);
  float*  gateL  = (float*)take((size_t)N_NODES * 32 * 4);
  float*  gateR  = (float*)take((size_t)N_NODES * 32 * 4);
  float*  dpos   = (float*)take((size_t)N_NODES * 3 * 4);
  __bf16* wLw1  = (__bf16*)take(128 * 128 * 2);
  __bf16* wLw2  = (__bf16*)take(128 * 128 * 2);
  __bf16* wRw1  = (__bf16*)take(128 * 128 * 2);
  __bf16* wRw2  = (__bf16*)take(128 * 128 * 2);
  __bf16* wBond = (__bf16*)take(128 * 128 * 2);
  __bf16* wNb   = (__bf16*)take(128 * 256 * 2);
  __bf16* wInt  = (__bf16*)take(128 * 128 * 2);
  __bf16* g1e   = (__bf16*)take(32 * 128 * 2);
  __bf16* g1l   = (__bf16*)take(32 * 128 * 2);
  __bf16* g1r   = (__bf16*)take(32 * 128 * 2);
  __bf16* sw1h  = (__bf16*)take(128 * 128 * 2);

  auto cvt = [&](const float* src, __bf16* dst, int rows, int srcLd, int colOff, int cols) {
    int n = rows * cols;
    cvt_kernel<<<(n + 255) / 256, 256, 0, stream>>>(src, dst, rows, srcLd, colOff, cols);
  };
  cvt(left_w1, wLw1, 128, 128, 0, 128);
  cvt(left_w2, wLw2, 128, 128, 0, 128);
  cvt(right_w1, wRw1, 128, 128, 0, 128);
  cvt(right_w2, wRw2, 128, 128, 0, 128);
  cvt(bond_lin_w, wBond, 128, 128, 0, 128);
  cvt(nodeb_lin_w, wNb, 128, 256, 0, 256);
  cvt(inter_w1, wInt, 128, 128, 0, 128);
  cvt(gate_w1, g1e, 32, 386, 0, 128);
  cvt(gate_w1, g1l, 32, 386, 128, 128);
  cvt(gate_w1, g1r, 32, 386, 256, 128);
  cvt(scale_w1, sw1h, 128, 131, 0, 128);

  hipMemsetAsync(dpos, 0, (size_t)N_NODES * 3 * 4, stream);

  int nodeBlocks = (N_NODES + 63) / 64;
  node_pre_kernel<<<nodeBlocks, 128, 0, stream>>>(
      h_node, node_extra, wLw1, left_b1, wLw2, left_b2, wRw1, right_b1, wRw2,
      right_b2, wNb, g1l, g1r, gate_w1, gate_b1, nodebL, nodebR, gateL, gateR);

  edge_kernel<<<N_EDGES / 64, 128, 0, stream>>>(
      h_edge, edge_index, relative_vec, distance, wBond, wInt, g1e, inter_b1,
      inter_w2, inter_b2, gate_w2, gate_b2, nodebL, nodebR, gateL, gateR, dpos);

  finish_kernel<<<nodeBlocks, 128, 0, stream>>>(
      h_node, node_extra, dpos, sw1h, scale_w1, scale_b1, scale_w2, scale_b2, out);
}
